// Conv2dShareQ_83992380440612
// MI455X (gfx1250) — compile-verified, matrix operand order checked by reading, not run on hardware
//
#include <hip/hip_runtime.h>
#include <hip/hip_bf16.h>

typedef __attribute__((ext_vector_type(16))) _Float16     v16h;
typedef __attribute__((ext_vector_type(8)))  float        v8f;
typedef __attribute__((ext_vector_type(8)))  unsigned int v8u;

#define CIN    256
#define COUT   512
#define HP     58                 // padded H/W
#define PLANE  (HP * HP)          // 3364
#define HW     3136               // 56*56
#define MT     128                // Cout rows per workgroup
#define SASTR  72                 // LDS stride (halfs) for A rows, 144B
#define SBSTR  72                 // LDS stride (halfs) for B rows
#define NWQ    (9 * COUT * CIN)   // dequantized weight count
#define NSTAGE 36                 // 9 taps x 4 ic-panels of 64

// ---------------------------------------------------------------------------
// Kernel 1: zero-pad + convert x to f16: xpad[b][ic][58][58]
// ---------------------------------------------------------------------------
__global__ void pad_convert_x(const float* __restrict__ x,
                              _Float16* __restrict__ xpad, int total) {
  int i = blockIdx.x * blockDim.x + threadIdx.x;
  if (i >= total) return;
  int plane = i / PLANE;
  int rem   = i - plane * PLANE;
  int ih = rem / HP;
  int iw = rem - ih * HP;
  _Float16 v = (_Float16)0.f;
  if (ih >= 1 && ih <= 56 && iw >= 1 && iw <= 56)
    v = (_Float16)x[(size_t)plane * HW + (ih - 1) * 56 + (iw - 1)];
  xpad[i] = v;
}

// ---------------------------------------------------------------------------
// Kernel 2: dequantize codebook weights -> f16, layout [tap=kh*3+kw][oc][ic]
// ---------------------------------------------------------------------------
__global__ void dequant_weights(const float* __restrict__ centroids,
                                const int* __restrict__ labels,
                                _Float16* __restrict__ wq2, int n) {
  int i = blockIdx.x * blockDim.x + threadIdx.x;
  if (i >= n) return;
  int t9  = i / (COUT * CIN);
  int rem = i - t9 * (COUT * CIN);
  int oc  = rem / CIN;
  int ic  = rem - oc * CIN;
  wq2[i] = (_Float16)centroids[labels[(oc * CIN + ic) * 9 + t9] & 15];
}

// ---------------------------------------------------------------------------
// Kernel 3: conv as 9 shifted GEMMs via v_wmma_f32_16x16x32_f16, with
// double-buffered LDS stages fed by GLOBAL_LOAD_ASYNC_TO_LDS_B128 (ASYNCcnt).
// Grid: (49 spatial 8x8 tiles, 4 M-tiles, 16 batches). Block: 256 = 8 waves.
// ---------------------------------------------------------------------------
__global__ __launch_bounds__(256)
void conv_wmma(const _Float16* __restrict__ xpad,
               const _Float16* __restrict__ wq2,
               const float* __restrict__ bias,
               float* __restrict__ out) {
  __shared__ _Float16 sA[2][MT * SASTR];  // weights: [oc][ic]
  __shared__ _Float16 sB[2][64 * SBSTR];  // input:   [pos][ic]

  const int sp = blockIdx.x;
  const int ty = sp / 7, tx = sp - ty * 7;
  const int h0 = ty * 8, w0 = tx * 8;
  const int m0 = blockIdx.y * MT;
  const int b  = blockIdx.z;

  const int tid  = threadIdx.x;
  const int wave = tid >> 5;
  const int lane = tid & 31;
  const int lrow = lane & 15;
  const int hi   = lane >> 4;

  // B-staging assignment: position p, 16-ic segment
  const int pB    = tid & 63;
  const int icseg = tid >> 6;
  const int rB    = pB >> 3, cB = pB & 7;

  // A-staging assignment: row r (0..127), 32-half segment; LDS byte offsets
  const int rA   = tid >> 1;
  const int segA = tid & 1;
  const unsigned int ldsA[2] = {
      (unsigned int)(unsigned long long)&sA[0][rA * SASTR + segA * 32],
      (unsigned int)(unsigned long long)&sA[1][rA * SASTR + segA * 32]};

  v8f acc[4] = {};

  const _Float16* xb = xpad + (size_t)b * CIN * PLANE;

  // ---- issue stage s into LDS buffer `buf` (async A DMA + B gather) ----
  auto issue = [&](int s, int buf) {
    const int t9  = s >> 2;
    const int ic0 = (s & 3) << 6;
    const int kh  = t9 / 3;
    const int kw  = t9 - kh * 3;

    // A tile: 128 oc-rows x 64 ic, DMA'd straight into LDS (64B/thread)
    unsigned long long ga = (unsigned long long)(const void*)(
        wq2 + ((size_t)t9 * COUT + m0 + rA) * CIN + ic0 + segA * 32);
    unsigned int la = ldsA[buf];
    asm volatile("global_load_async_to_lds_b128 %0, %1, off"
                 :: "v"(la), "v"(ga) : "memory");
    asm volatile("global_load_async_to_lds_b128 %0, %1, off offset:16"
                 :: "v"(la), "v"(ga) : "memory");
    asm volatile("global_load_async_to_lds_b128 %0, %1, off offset:32"
                 :: "v"(la), "v"(ga) : "memory");
    asm volatile("global_load_async_to_lds_b128 %0, %1, off offset:48"
                 :: "v"(la), "v"(ga) : "memory");

    // B tile: 64 pos x 64 ic; 16 ic per thread, fixed position
    const _Float16* src = xb + (size_t)(h0 + rB + kh) * HP + (w0 + cB + kw)
                        + (size_t)(ic0 + icseg * 16) * PLANE;
    _Float16* dst = &sB[buf][pB * SBSTR + icseg * 16];
#pragma unroll
    for (int i = 0; i < 16; ++i)
      dst[i] = src[(size_t)i * PLANE];
    if (ic0 < 192)  // stays inside this tap's valid ic range
      __builtin_prefetch(src + (size_t)64 * PLANE, 0, 0);
  };

  // ---- compute stage from LDS buffer `buf`: 2 ksubs x 4 nsubs = 8 WMMA ----
  auto compute = [&](int buf) {
#pragma unroll
    for (int ksub = 0; ksub < 2; ++ksub) {
      const int kb = ksub * 32 + hi * 8;

      const uint4* ap = reinterpret_cast<const uint4*>(
          &sA[buf][(wave * 16 + lrow) * SASTR + kb]);
      uint4 a_lo = ap[0];
      uint4 a_hi = ap[2];
      v8u ar;
      ar[0] = a_lo.x; ar[1] = a_lo.y; ar[2] = a_lo.z; ar[3] = a_lo.w;
      ar[4] = a_hi.x; ar[5] = a_hi.y; ar[6] = a_hi.z; ar[7] = a_hi.w;
      v16h afrag = __builtin_bit_cast(v16h, ar);

#pragma unroll
      for (int ns = 0; ns < 4; ++ns) {
        const uint4* bp = reinterpret_cast<const uint4*>(
            &sB[buf][(ns * 16 + lrow) * SBSTR + kb]);
        uint4 b_lo = bp[0];
        uint4 b_hi = bp[2];
        v8u br;
        br[0] = b_lo.x; br[1] = b_lo.y; br[2] = b_lo.z; br[3] = b_lo.w;
        br[4] = b_hi.x; br[5] = b_hi.y; br[6] = b_hi.z; br[7] = b_hi.w;
        v16h bfrag = __builtin_bit_cast(v16h, br);
        acc[ns] = __builtin_amdgcn_wmma_f32_16x16x32_f16(
            false, afrag, false, bfrag, (short)0, acc[ns], false, false);
      }
    }
  };

  // ---- software pipeline: DMA of stage s+1 overlaps WMMAs of stage s ----
  issue(0, 0);
  asm volatile("s_wait_asynccnt 0x0" ::: "memory");
  __syncthreads();

  for (int s = 0; s < NSTAGE; ++s) {
    const int cur = s & 1;
    if (s + 1 < NSTAGE) issue(s + 1, cur ^ 1);
    compute(cur);
    asm volatile("s_wait_asynccnt 0x0" ::: "memory");  // stage s+1 DMA done
    __syncthreads();  // all waves done writing buf^1 / reading buf
  }

  // ---- epilogue: bias + store ----
  const size_t halfOut = (size_t)16 * 256 * HW;
#pragma unroll
  for (int ns = 0; ns < 4; ++ns) {
    int p = ns * 16 + lrow;
    int n = (h0 + (p >> 3)) * 56 + w0 + (p & 7);
#pragma unroll
    for (int j = 0; j < 8; ++j) {
      int row = j + 8 * hi;
      int oc  = m0 + wave * 16 + row;
      float v = acc[ns][j] + bias[oc];
      size_t off = (oc < 256)
          ? ((size_t)b * 256 + oc) * (size_t)HW
          : halfOut + ((size_t)b * 256 + (oc - 256)) * (size_t)HW;
      out[off + n] = v;
    }
  }
}

// ---------------------------------------------------------------------------
extern "C" void kernel_launch(void* const* d_in, const int* in_sizes, int n_in,
                              void* d_out, int out_size, void* d_ws, size_t ws_size,
                              hipStream_t stream) {
  (void)in_sizes; (void)n_in; (void)out_size; (void)ws_size;
  const float* x         = (const float*)d_in[0];
  const float* centroids = (const float*)d_in[1];
  const int*   labels    = (const int*)d_in[2];
  const float* bias      = (const float*)d_in[3];
  float*       out       = (float*)d_out;

  // ws layout: [wq2: 2.36 MB][xpad: 27.6 MB]
  _Float16* wq2  = (_Float16*)d_ws;
  _Float16* xpad = (_Float16*)((char*)d_ws + (size_t)NWQ * sizeof(_Float16));

  dequant_weights<<<(NWQ + 255) / 256, 256, 0, stream>>>(centroids, labels,
                                                         wq2, NWQ);
  const int npad = 16 * CIN * PLANE;
  pad_convert_x<<<(npad + 255) / 256, 256, 0, stream>>>(x, xpad, npad);

  dim3 grid(49, COUT / MT, 16);
  conv_wmma<<<grid, 256, 0, stream>>>(xpad, wq2, bias, out);
}
